// DGA_75591424410317
// MI455X (gfx1250) — matrix operand order repaired and checked by reading, not compiled, not run on hardware
//
#include <hip/hip_runtime.h>
#include <hip/hip_bf16.h>

// ---------------------------------------------------------------------------
// GNN layer for MI455X (gfx1250, wave32). GEMMs via V_WMMA_F32_16X16X32_BF16,
// register-blocked: one wave owns a 16-row strip and all M/16 column tiles,
// so the A fragment is loaded once per k-step and reused across 8 WMMAs.
// Edge aggregation fused into a single pass (s_g1 = s_all - s_g0).
// Workspace usage ~360 MB.
// ---------------------------------------------------------------------------

typedef __attribute__((ext_vector_type(16))) __bf16 bf16x16;
typedef __attribute__((ext_vector_type(8)))  __bf16 bf16x8;
typedef __attribute__((ext_vector_type(8)))  float  f32x8;

#define HD 128   // hidden dim == input dim

// ---------------- utility kernels ----------------

__global__ void k_zero(float* __restrict__ p, size_t n) {
  size_t i = (size_t)blockIdx.x * blockDim.x + threadIdx.x;
  size_t st = (size_t)gridDim.x * blockDim.x;
  for (; i < n; i += st) p[i] = 0.f;
}

__global__ void k_cvt_bf16(const float* __restrict__ s, __bf16* __restrict__ d, size_t n) {
  size_t i = (size_t)blockIdx.x * blockDim.x + threadIdx.x;
  if (i < n) d[i] = (__bf16)s[i];
}

// ---------------- WMMA GEMM:  Y[N,M] (+)= A[N,128] x W[M,128]^T (+ bias) ----
// One wave computes a full 16xM strip: M/16 accumulator tiles, K=128 -> 4
// k-steps, A fragment loaded once per k-step, 8 (or 4) WMMAs per k-step.
// A fragment (16-bit A 16x32): lane(half=l>>4, r=l&15) holds row r,
//   elems 0..15 = A[r][k0 + half*16 + i].
// B fragment (16-bit B 32x16): lane holds col r, elems = W[col][k0+half*16+i].
// C/D f32 16x16: lane holds col r, rows half*8 + v (v = vgpr index).

template <int M>
__global__ __launch_bounds__(128) void k_wmma_gemm(
    const __bf16* __restrict__ A, const __bf16* __restrict__ W,
    const float* __restrict__ bias, float* __restrict__ Y,
    int rowTiles, int accumulate)
{
  constexpr int CT = M / 16;                     // column tiles per strip
  const int lane  = threadIdx.x & 31;
  const int wave  = threadIdx.x >> 5;
  const int strip = blockIdx.x * 4 + wave;       // 16-row strip index
  if (strip >= rowTiles) return;                 // whole wave exits: EXEC stays full
  const int half = lane >> 4;
  const int r    = lane & 15;
  const int rowBase = strip << 4;

  const __bf16* aBase = A + (size_t)(rowBase + r) * HD + half * 16;
  const __bf16* wBase = W + (size_t)r * HD + half * 16;   // + ct*16*HD per tile

  f32x8 acc[CT];
#pragma unroll
  for (int ct = 0; ct < CT; ++ct)
    acc[ct] = (f32x8){0.f, 0.f, 0.f, 0.f, 0.f, 0.f, 0.f, 0.f};
  if (accumulate) {
#pragma unroll
    for (int ct = 0; ct < CT; ++ct)
#pragma unroll
      for (int v = 0; v < 8; ++v)
        acc[ct][v] = Y[(size_t)(rowBase + half * 8 + v) * M + ct * 16 + r];
  }

#pragma unroll
  for (int k0 = 0; k0 < HD; k0 += 32) {
    bf16x8 alo = *(const bf16x8*)(aBase + k0);
    bf16x8 ahi = *(const bf16x8*)(aBase + k0 + 8);
    bf16x16 afrag = __builtin_shufflevector(alo, ahi,
        0,1,2,3,4,5,6,7,8,9,10,11,12,13,14,15);
#pragma unroll
    for (int ct = 0; ct < CT; ++ct) {
      bf16x16 bfrag = *(const bf16x16*)(wBase + (size_t)ct * 16 * HD + k0);
      acc[ct] = __builtin_amdgcn_wmma_f32_16x16x32_bf16(
          false, afrag, false, bfrag, (short)0, acc[ct], false, false);
    }
  }

#pragma unroll
  for (int ct = 0; ct < CT; ++ct) {
    const float bv = bias ? bias[ct * 16 + r] : 0.f;
#pragma unroll
    for (int v = 0; v < 8; ++v)
      Y[(size_t)(rowBase + half * 8 + v) * M + ct * 16 + r] = acc[ct][v] + bv;
  }
}

// ---------------- BN statistics (per-column sum / sumsq) -------------------

__global__ __launch_bounds__(HD) void k_col_stats(
    const float* __restrict__ Z, int nRows,
    float* __restrict__ sum, float* __restrict__ ssq)
{
  const int col = threadIdx.x;                  // blockDim.x == 128
  const int rowsPer = (nRows + gridDim.x - 1) / gridDim.x;
  int r0 = blockIdx.x * rowsPer;
  int r1 = r0 + rowsPer; if (r1 > nRows) r1 = nRows;
  float s = 0.f, q = 0.f;
  for (int rr = r0; rr < r1; ++rr) {
    float v = Z[(size_t)rr * HD + col];
    s += v; q += v * v;
  }
  atomicAdd(&sum[col], s);
  atomicAdd(&ssq[col], q);
}

// BN (training stats) -> ReLU, write f32 + bf16 (emb layers; Linear bias absorbed)
__global__ void k_bn_relu(const float* __restrict__ Z,
                          const float* __restrict__ g, const float* __restrict__ be,
                          const float* __restrict__ sum, const float* __restrict__ ssq,
                          int nRows, float* __restrict__ fo, __bf16* __restrict__ bo)
{
  size_t i = (size_t)blockIdx.x * blockDim.x + threadIdx.x;
  if (i >= (size_t)nRows * HD) return;
  int col = (int)(i & (HD - 1));
  float inv = 1.f / (float)nRows;
  float m = sum[col] * inv;
  float var = ssq[col] * inv - m * m;
  float y = g[col] * (Z[i] - m) * rsqrtf(var + 1e-5f) + be[col];
  y = fmaxf(y, 0.f);
  fo[i] = y;
  bo[i] = (__bf16)y;
}

// conv path: z = relu(z + bias)  (in place, before BN)
__global__ void k_relu_bias(float* __restrict__ Z, const float* __restrict__ bias, size_t total)
{
  size_t i = (size_t)blockIdx.x * blockDim.x + threadIdx.x;
  if (i >= total) return;
  Z[i] = fmaxf(Z[i] + bias[i & (HD - 1)], 0.f);
}

// conv path: y = BN(z) (+ addsrc); write f32 and/or bf16 (nullable)
__global__ void k_bn_out(const float* __restrict__ Z,
                         const float* __restrict__ g, const float* __restrict__ be,
                         const float* __restrict__ sum, const float* __restrict__ ssq,
                         int nRows, const float* __restrict__ addsrc,
                         float* __restrict__ fo, __bf16* __restrict__ bo)
{
  size_t i = (size_t)blockIdx.x * blockDim.x + threadIdx.x;
  if (i >= (size_t)nRows * HD) return;
  int col = (int)(i & (HD - 1));
  float inv = 1.f / (float)nRows;
  float m = sum[col] * inv;
  float var = ssq[col] * inv - m * m;
  float y = g[col] * (Z[i] - m) * rsqrtf(var + 1e-5f) + be[col];
  if (addsrc) y += addsrc[i];
  if (fo) fo[i] = y;
  if (bo) bo[i] = (__bf16)y;
}

// ---------------- grouping head: C==2 argmax -> per-node w0 indicator ------

__global__ void k_group_w0(const float* __restrict__ h, const float* __restrict__ Wc,
                           const float* __restrict__ bc, float* __restrict__ w0node, int nRows)
{
  int lane = threadIdx.x & 31;
  long n = ((long)blockIdx.x * blockDim.x + threadIdx.x) >> 5;
  if (n >= nRows) return;
  const float* hp = h + (size_t)n * HD;
  float d = 0.f;
#pragma unroll
  for (int j = 0; j < 4; ++j) {
    int k = lane * 4 + j;
    d += hp[k] * (Wc[HD + k] - Wc[k]);          // logit1 - logit0
  }
#pragma unroll
  for (int off = 16; off; off >>= 1) d += __shfl_xor(d, off, 32);
  if (lane == 0) {
    d += bc[1] - bc[0];
    w0node[n] = (d > 0.f) ? 0.f : 1.f;          // argmax==0 <=> l1 <= l0
  }
}

// ---------------- fused edge aggregation (one pass, warp per edge) ---------
// s_all += h[src]; s_g0 += w0*h[src]; deg += 1; dw0 += w0
// h fits in the 192MB L2 -> gather runs at L2 bandwidth. The w0 branch is
// wave-uniform (one edge per wave) so there is no divergence.

__global__ __launch_bounds__(256) void k_edge_agg(
    const int* __restrict__ src, const int* __restrict__ dst,
    const float* __restrict__ h, const float* __restrict__ w0node,
    float* __restrict__ s_all, float* __restrict__ s_g0,
    float* __restrict__ deg, float* __restrict__ dw0, long E)
{
  int lane = threadIdx.x & 31;
  long e = ((long)blockIdx.x * blockDim.x + threadIdx.x) >> 5;
  if (e >= E) return;
  int s = src[e], t = dst[e];
  float w = w0node[s];
  float4 v = ((const float4*)(h + (size_t)s * HD))[lane];
  float* sa = s_all + (size_t)t * HD + lane * 4;
  atomicAdd(sa + 0, v.x); atomicAdd(sa + 1, v.y);
  atomicAdd(sa + 2, v.z); atomicAdd(sa + 3, v.w);
  if (w != 0.f) {
    float* s0 = s_g0 + (size_t)t * HD + lane * 4;
    atomicAdd(s0 + 0, v.x); atomicAdd(s0 + 1, v.y);
    atomicAdd(s0 + 2, v.z); atomicAdd(s0 + 3, v.w);
  }
  if (lane == 0) { atomicAdd(deg + t, 1.f); atomicAdd(dw0 + t, w); }
}

// normalize the three neighbor aggregates, emit bf16 A-operands directly
__global__ void k_neigh_div(const float* __restrict__ s_all, const float* __restrict__ s_g0,
                            const float* __restrict__ deg, const float* __restrict__ dw0,
                            __bf16* __restrict__ nb_all, __bf16* __restrict__ nb_g0,
                            __bf16* __restrict__ nb_g1, int nRows)
{
  size_t i = (size_t)blockIdx.x * blockDim.x + threadIdx.x;
  if (i >= (size_t)nRows * HD) return;
  int n = (int)(i >> 7);
  float d  = deg[n], d0 = dw0[n];
  float sa = s_all[i], s0 = s_g0[i];
  float na = (d > 0.f) ? sa / fmaxf(d, 1.f) : 0.f;
  float n0 = s0 / (d0 + 1e-8f);
  float n1 = (sa - s0) / ((d - d0) + 1e-8f);
  nb_all[i] = (__bf16)na;
  nb_g0[i]  = (__bf16)n0;
  nb_g1[i]  = (__bf16)n1;
}

// fused = tanh(yf) * tanh(yx) -> bf16
__global__ void k_tanh_mul(const float* __restrict__ yf, const float* __restrict__ yx,
                           __bf16* __restrict__ fused, size_t total)
{
  size_t i = (size_t)blockIdx.x * blockDim.x + threadIdx.x;
  if (i >= total) return;
  fused[i] = (__bf16)(tanhf(yf[i]) * tanhf(yx[i]));
}

// relu -> bf16 (after Wo1 GEMM, bias already folded into GEMM epilogue)
__global__ void k_relu_bf(const float* __restrict__ z, __bf16* __restrict__ o, size_t total)
{
  size_t i = (size_t)blockIdx.x * blockDim.x + threadIdx.x;
  if (i >= total) return;
  o[i] = (__bf16)fmaxf(z[i], 0.f);
}

// final: out[n, 0..1] = t1[n,:64] . Wo2^T + bo2   (warp per node; M=2 too small for WMMA)
__global__ void k_final(const __bf16* __restrict__ t1, const float* __restrict__ Wo2,
                        const float* __restrict__ bo2, float* __restrict__ out, int nRows)
{
  int lane = threadIdx.x & 31;
  long n = ((long)blockIdx.x * blockDim.x + threadIdx.x) >> 5;
  if (n >= nRows) return;
  const __bf16* tp = t1 + (size_t)n * 64;
  float a0 = 0.f, a1 = 0.f;
#pragma unroll
  for (int j = 0; j < 2; ++j) {
    int k = lane * 2 + j;
    float tv = (float)tp[k];
    a0 += tv * Wo2[k];
    a1 += tv * Wo2[64 + k];
  }
#pragma unroll
  for (int off = 16; off; off >>= 1) { a0 += __shfl_xor(a0, off, 32); a1 += __shfl_xor(a1, off, 32); }
  if (lane == 0) { out[n * 2 + 0] = a0 + bo2[0]; out[n * 2 + 1] = a1 + bo2[1]; }
}

// ---------------------------------------------------------------------------

extern "C" void kernel_launch(void* const* d_in, const int* in_sizes, int n_in,
                              void* d_out, int out_size, void* d_ws, size_t ws_size,
                              hipStream_t stream)
{
  const int  N  = in_sizes[0] / HD;        // 100000 (multiple of 16)
  const long E  = in_sizes[1];             // 1600000
  const size_t NHf = (size_t)N * HD;

  // ---- inputs ----
  const float* x        = (const float*)d_in[0];
  const int*   esrc     = (const int*)  d_in[1];
  const int*   edst     = (const int*)  d_in[2];
  const float* W1       = (const float*)d_in[3];
  const float* g1       = (const float*)d_in[5];
  const float* be1      = (const float*)d_in[6];
  const float* W2       = (const float*)d_in[7];
  const float* g2       = (const float*)d_in[9];
  const float* be2      = (const float*)d_in[10];
  const float* Wc       = (const float*)d_in[11];
  const float* bc       = (const float*)d_in[12];
  const float* Ws_all   = (const float*)d_in[13];
  const float* Wn_all   = (const float*)d_in[14];
  const float* bias_all = (const float*)d_in[15];
  const float* gn_all   = (const float*)d_in[16];
  const float* bn_all   = (const float*)d_in[17];
  const float* Wn_g0    = (const float*)d_in[19];
  const float* bias_g0  = (const float*)d_in[20];
  const float* gn_g0    = (const float*)d_in[21];
  const float* bn_g0    = (const float*)d_in[22];
  const float* Wn_g1    = (const float*)d_in[24];
  const float* bias_g1  = (const float*)d_in[25];
  const float* gn_g1    = (const float*)d_in[26];
  const float* bn_g1    = (const float*)d_in[27];
  const float* Wf       = (const float*)d_in[28];
  const float* bf       = (const float*)d_in[29];
  const float* Wx       = (const float*)d_in[30];
  const float* bx       = (const float*)d_in[31];
  const float* Wo1      = (const float*)d_in[32];
  const float* bo1      = (const float*)d_in[33];
  const float* Wo2      = (const float*)d_in[34];
  const float* bo2      = (const float*)d_in[35];
  float* out = (float*)d_out;

  // ---- workspace carve-up ----
  char* ws = (char*)d_ws;
  size_t off = 0;
  auto carve = [&](size_t bytes) { size_t o = off; off = (off + bytes + 255) & ~(size_t)255; return o; };
  float*  F_h    = (float*)(ws + carve(NHf * 4));   // h (f32, emb output)
  float*  F_y0   = (float*)(ws + carve(NHf * 4));   // GEMM scratch 0
  float*  F_y1   = (float*)(ws + carve(NHf * 4));   // GEMM scratch 1
  float*  F_sall = (float*)(ws + carve(NHf * 4));   // sum_all -> later h_g0 f32
  float*  F_sg0  = (float*)(ws + carve(NHf * 4));   // sum_g0
  __bf16* B_a    = (__bf16*)(ws + carve(NHf * 2));  // x_bf -> h_bf
  __bf16* B_1    = (__bf16*)(ws + carve(NHf * 2));  // nb_all -> hall_bf
  __bf16* B_2    = (__bf16*)(ws + carve(NHf * 2));  // nb_g0 -> hg_bf -> t1_bf
  __bf16* B_3    = (__bf16*)(ws + carve(NHf * 2));  // nb_g1 -> fused_bf
  float*  F_deg  = (float*)(ws + carve((size_t)N * 4));
  float*  F_dw0  = (float*)(ws + carve((size_t)N * 4));
  float*  F_w0n  = (float*)(ws + carve((size_t)N * 4));
  float*  F_sum  = (float*)(ws + carve(256 * 4));   // [0:128)=sum, [128:256)=sumsq
  __bf16* B_w    = (__bf16*)(ws + carve((size_t)9 * HD * HD * 2));

  auto wslot = [&](int i) { return B_w + (size_t)i * HD * HD; };

  const int EW = 256;
  auto elems = [](size_t n, int b) { return (int)((n + b - 1) / b); };

  // ---- weight + input conversions to bf16 ----
  k_cvt_bf16<<<elems(NHf, EW), EW, 0, stream>>>(x, B_a, NHf);
  k_cvt_bf16<<<64, EW, 0, stream>>>(W1,     wslot(0), HD * HD);
  k_cvt_bf16<<<64, EW, 0, stream>>>(W2,     wslot(1), HD * HD);
  k_cvt_bf16<<<64, EW, 0, stream>>>(Wn_all, wslot(2), HD * HD);
  k_cvt_bf16<<<64, EW, 0, stream>>>(Ws_all, wslot(3), HD * HD);
  k_cvt_bf16<<<64, EW, 0, stream>>>(Wn_g0,  wslot(4), HD * HD);
  k_cvt_bf16<<<64, EW, 0, stream>>>(Wn_g1,  wslot(5), HD * HD);
  k_cvt_bf16<<<64, EW, 0, stream>>>(Wf,     wslot(6), HD * HD);
  k_cvt_bf16<<<64, EW, 0, stream>>>(Wx,     wslot(7), HD * HD);
  k_cvt_bf16<<<32, EW, 0, stream>>>(Wo1,    wslot(8), 64 * HD);

  const int rowTiles = N / 16;
  const int gemmBlocks = (rowTiles + 3) / 4;
  auto gemm = [&](const __bf16* A, const __bf16* Wb, const float* bias,
                  float* Y, int M, int acc) {
    if (M == 128)
      k_wmma_gemm<128><<<gemmBlocks, 128, 0, stream>>>(A, Wb, bias, Y, rowTiles, acc);
    else
      k_wmma_gemm<64><<<gemmBlocks, 128, 0, stream>>>(A, Wb, bias, Y, rowTiles, acc);
  };
  auto stats = [&](const float* Z) {
    k_zero<<<1, 256, 0, stream>>>(F_sum, 256);
    k_col_stats<<<256, HD, 0, stream>>>(Z, N, F_sum, F_sum + HD);
  };
  const int GE = elems(NHf, EW);   // elementwise grid over N*H

  // ---- emb layer 1 (BN absorbs b1) ----
  gemm(B_a, wslot(0), nullptr, F_y0, HD, 0);
  stats(F_y0);
  k_bn_relu<<<GE, EW, 0, stream>>>(F_y0, g1, be1, F_sum, F_sum + HD, N, F_h, B_a);
  // ---- emb layer 2 ----
  gemm(B_a, wslot(1), nullptr, F_y0, HD, 0);
  stats(F_y0);
  k_bn_relu<<<GE, EW, 0, stream>>>(F_y0, g2, be2, F_sum, F_sum + HD, N, F_h, B_a);

  // ---- grouping head ----
  k_group_w0<<<elems((size_t)N * 32, EW), EW, 0, stream>>>(F_h, Wc, bc, F_w0n, N);

  // ---- fused edge aggregation ----
  k_zero<<<2048, EW, 0, stream>>>(F_sall, NHf);
  k_zero<<<2048, EW, 0, stream>>>(F_sg0,  NHf);
  k_zero<<<256,  EW, 0, stream>>>(F_deg,  (size_t)N);
  k_zero<<<256,  EW, 0, stream>>>(F_dw0,  (size_t)N);
  k_edge_agg<<<elems((size_t)E * 32, EW), EW, 0, stream>>>(
      esrc, edst, F_h, F_w0n, F_sall, F_sg0, F_deg, F_dw0, E);
  k_neigh_div<<<GE, EW, 0, stream>>>(F_sall, F_sg0, F_deg, F_dw0, B_1, B_2, B_3, N);

  // ---- conv "all": neigh@Wn + h@Ws, relu(+bias), BN -> hall_bf in B_1 ----
  gemm(B_1, wslot(2), nullptr, F_y0, HD, 0);
  gemm(B_a, wslot(3), nullptr, F_y0, HD, 1);        // accumulate self term
  k_relu_bias<<<GE, EW, 0, stream>>>(F_y0, bias_all, NHf);
  stats(F_y0);
  k_bn_out<<<GE, EW, 0, stream>>>(F_y0, gn_all, bn_all, F_sum, F_sum + HD, N,
                                  nullptr, nullptr, B_1);
  // ---- conv g0 -> h_g0 f32 in F_sall ----
  gemm(B_2, wslot(4), nullptr, F_y0, HD, 0);
  k_relu_bias<<<GE, EW, 0, stream>>>(F_y0, bias_g0, NHf);
  stats(F_y0);
  k_bn_out<<<GE, EW, 0, stream>>>(F_y0, gn_g0, bn_g0, F_sum, F_sum + HD, N,
                                  nullptr, F_sall, nullptr);
  // ---- conv g1 -> (h_g0 + h_g1) bf16 in B_2 ----
  gemm(B_3, wslot(5), nullptr, F_y0, HD, 0);
  k_relu_bias<<<GE, EW, 0, stream>>>(F_y0, bias_g1, NHf);
  stats(F_y0);
  k_bn_out<<<GE, EW, 0, stream>>>(F_y0, gn_g1, bn_g1, F_sum, F_sum + HD, N,
                                  F_sall, nullptr, B_2);

  // ---- tanh attention fusion ----
  gemm(B_1, wslot(6), bf, F_y0, HD, 0);             // h_all @ Wf^T + bf
  gemm(B_2, wslot(7), bx, F_y1, HD, 0);             // (h_g0+h_g1) @ Wx^T + bx
  k_tanh_mul<<<GE, EW, 0, stream>>>(F_y0, F_y1, B_3, NHf);

  // ---- final MLP ----
  gemm(B_3, wslot(8), bo1, F_y0, 64, 0);            // [N,64] = fused @ Wo1^T + bo1
  k_relu_bf<<<elems((size_t)N * 64, EW), EW, 0, stream>>>(F_y0, B_2, (size_t)N * 64);
  k_final<<<elems((size_t)N * 32, EW), EW, 0, stream>>>(B_2, Wo2, bo2, out, N);
}
